// MultiHeadAttentionOne_17033840296536
// MI455X (gfx1250) — compile-verified, matrix operand order checked by reading, not run on hardware
//
#include <hip/hip_runtime.h>
#include <hip/hip_bf16.h>

typedef __attribute__((ext_vector_type(16))) _Float16 v16h;
typedef __attribute__((ext_vector_type(8)))  _Float16 v8h;
typedef __attribute__((ext_vector_type(8)))  float    v8f;
typedef __attribute__((ext_vector_type(4)))  int      v4i;

#define B_SZ   8
#define LQ_TOT 3600
#define LK_TOT 3600
#define DM     512
#define M_TOT  (B_SZ * LQ_TOT)   // 28800

#if __has_builtin(__builtin_amdgcn_global_load_async_to_lds_b128)
#define HAVE_ASYNC_LDS 1
#endif

typedef __attribute__((address_space(1))) v4i* as1_v4i;
typedef __attribute__((address_space(3))) v4i* as3_v4i;

static __device__ __forceinline__ void wait_asynccnt0() {
#if defined(HAVE_ASYNC_LDS)
#if __has_builtin(__builtin_amdgcn_s_wait_asynccnt)
    __builtin_amdgcn_s_wait_asynccnt(0);
#else
    asm volatile("s_wait_asynccnt 0x0" ::: "memory");
#endif
#endif
}

static __device__ __forceinline__ v8f v8f_zero() {
    v8f z = {0.f, 0.f, 0.f, 0.f, 0.f, 0.f, 0.f, 0.f};
    return z;
}

static __device__ __forceinline__ v16h mk_frag(v8h lo, v8h hi8) {
    union U { v16h v; v8h h[2]; } u;
    u.h[0] = lo; u.h[1] = hi8;
    return u.v;
}

// Load an A/B fragment for v_wmma_f32_16x16x32_f16.
// rowbase must point at (row/col selected by lane&15) + k offset; 'hi' = lane>>4.
// Layout: VGPR0-3 hold K = hi*8 + 0..7, VGPR4-7 hold K = 16 + hi*8 + 0..7.
static __device__ __forceinline__ v16h load_frag(const _Float16* rowbase, int hi) {
    const _Float16* p = rowbase + hi * 8;
    return mk_frag(*(const v8h*)p, *(const v8h*)(p + 16));
}

// ---------------------------------------------------------------------------
// Pack f32 -> f16 (flat)
// ---------------------------------------------------------------------------
__global__ void __launch_bounds__(256) pack_f16(const float* __restrict__ in,
                                                _Float16* __restrict__ out, int n4) {
    int i = blockIdx.x * 256 + threadIdx.x;
    if (i < n4) {
        float4 f = ((const float4*)in)[i];
        union { _Float16 h[4]; unsigned long long u; } u;
        u.h[0] = (_Float16)f.x; u.h[1] = (_Float16)f.y;
        u.h[2] = (_Float16)f.z; u.h[3] = (_Float16)f.w;
        ((unsigned long long*)out)[i] = u.u;
    }
}

// ---------------------------------------------------------------------------
// Transpose [B][512][3600] f32 -> [B*3600][512] f16 (32x32 LDS tiles)
// ---------------------------------------------------------------------------
__global__ void __launch_bounds__(256) transpose_pack(const float* __restrict__ in,
                                                      _Float16* __restrict__ out) {
    __shared__ float t[32][33];
    const int b  = blockIdx.z;
    const int l0 = blockIdx.x * 32;
    const int d0 = blockIdx.y * 32;
    const int tx = threadIdx.x & 31;
    const int ty = threadIdx.x >> 5;
    const size_t inb = (size_t)b * DM * LK_TOT;
#pragma unroll
    for (int i = 0; i < 4; ++i) {
        int d = d0 + ty + i * 8, l = l0 + tx;
        if (l < LK_TOT) t[ty + i * 8][tx] = in[inb + (size_t)d * LK_TOT + l];
    }
    __syncthreads();
#pragma unroll
    for (int i = 0; i < 4; ++i) {
        int l = l0 + ty + i * 8, d = d0 + tx;
        if (l < LK_TOT)
            out[((size_t)b * LK_TOT + l) * DM + d] = (_Float16)t[tx][ty + i * 8];
    }
}

// ---------------------------------------------------------------------------
// NT GEMM: C[M,512] = A[M,512] * W[512,512]^T, f16 in, WMMA f32 accumulate.
// Block = 256 threads (8 waves). Block tile 128x256, wave tile 32x128.
// MODE 0: store f16 row-major.
// MODE 1: store f16 transposed into [512][M_TOT] (for V).
// MODE 2: store f32 row-major + bias + residual (fc layer).
// ---------------------------------------------------------------------------
template <int MODE>
__global__ void __launch_bounds__(256)
gemm512(const _Float16* __restrict__ A, const _Float16* __restrict__ W,
        _Float16* __restrict__ o16, float* __restrict__ o32,
        const float* __restrict__ bias, const float* __restrict__ resid) {
    const int tid  = threadIdx.x;
    const int lane = tid & 31;
    const int w    = tid >> 5;
    const int ln   = lane & 15;
    const int hi   = lane >> 4;
    const int m0   = blockIdx.x * 128 + (w & 3) * 32;
    const int n0   = blockIdx.y * 256 + (w >> 2) * 128;

    v8f acc[2][8];
#pragma unroll
    for (int mi = 0; mi < 2; ++mi)
#pragma unroll
        for (int f = 0; f < 8; ++f) acc[mi][f] = v8f_zero();

    const _Float16* a0 = A + (size_t)(m0 + ln) * DM;
    const _Float16* a1 = A + (size_t)(m0 + 16 + ln) * DM;

    for (int k = 0; k < DM; k += 32) {
        v16h fa0 = load_frag(a0 + k, hi);
        v16h fa1 = load_frag(a1 + k, hi);
#pragma unroll
        for (int f = 0; f < 8; ++f) {
            const _Float16* br = W + (size_t)(n0 + 16 * f + ln) * DM;
            v16h fb = load_frag(br + k, hi);
            acc[0][f] = __builtin_amdgcn_wmma_f32_16x16x32_f16(
                false, fa0, false, fb, (short)0, acc[0][f], false, false);
            acc[1][f] = __builtin_amdgcn_wmma_f32_16x16x32_f16(
                false, fa1, false, fb, (short)0, acc[1][f], false, false);
        }
    }

#pragma unroll
    for (int mi = 0; mi < 2; ++mi) {
        const int r0 = m0 + 16 * mi + hi * 8;
#pragma unroll
        for (int f = 0; f < 8; ++f) {
            const int col = n0 + 16 * f + ln;
            if (MODE == 1) {
                v8h ph;
#pragma unroll
                for (int g = 0; g < 8; ++g) ph[g] = (_Float16)acc[mi][f][g];
                *(v8h*)(o16 + (size_t)col * M_TOT + r0) = ph;
            } else if (MODE == 0) {
#pragma unroll
                for (int g = 0; g < 8; ++g)
                    o16[(size_t)(r0 + g) * DM + col] = (_Float16)acc[mi][f][g];
            } else {
                const float bv = bias[col];
#pragma unroll
                for (int g = 0; g < 8; ++g) {
                    size_t idx = (size_t)(r0 + g) * DM + col;
                    o32[idx] = acc[mi][f][g] + bv + resid[idx];
                }
            }
        }
    }
}

// ---------------------------------------------------------------------------
// Flash attention: block = 32 q-rows, 8 waves, 64-key tiles, online softmax.
// qp: [B*Lq,512] f16  kp: [B*Lk,512] f16  vpT: [512][B*Lk] f16
// attn out: [B*Lq,512] f16
// ---------------------------------------------------------------------------
__global__ void __launch_bounds__(256)
flash_attn(const _Float16* __restrict__ qp, const _Float16* __restrict__ kp,
           const _Float16* __restrict__ vpT, _Float16* __restrict__ attn) {
    __shared__ __align__(16) _Float16 sQ[32][520];
    __shared__ float                  sS[32][68];
    __shared__ __align__(16) _Float16 sP[32][72];
    __shared__ float sM[32], sL[32], sAlpha[32];

    const int tid  = threadIdx.x;
    const int lane = tid & 31;
    const int w    = tid >> 5;
    const int ln   = lane & 15;
    const int hi   = lane >> 4;
    const int b    = blockIdx.y;
    const int q0   = blockIdx.x * 32;
    const size_t bL = (size_t)b * LQ_TOT;
    const float scale = 0.044194173824159216f;  // 1/sqrt(512)

    // Stage Q tile into LDS (async DMA path when available; clamp ragged rows).
    for (int c = tid; c < 32 * 64; c += 256) {
        int r  = c >> 6;
        int k8 = (c & 63) * 8;
        int gr = min(q0 + r, LQ_TOT - 1);
        const _Float16* src = qp + (bL + gr) * (size_t)DM + k8;
#if defined(HAVE_ASYNC_LDS)
        __builtin_amdgcn_global_load_async_to_lds_b128(
            (as1_v4i)src, (as3_v4i)&sQ[r][k8], 0, 0);
#else
        *(v8h*)(&sQ[r][k8]) = *(const v8h*)src;
#endif
    }
    if (tid < 32) { sM[tid] = -1e30f; sL[tid] = 0.f; }
    wait_asynccnt0();

    v8f o[2][4];
#pragma unroll
    for (int mi = 0; mi < 2; ++mi)
#pragma unroll
        for (int ni = 0; ni < 4; ++ni) o[mi][ni] = v8f_zero();

    __syncthreads();

    const int mb = w & 1;   // S-frag row block (16 rows)
    const int nb = w >> 1;  // S-frag col block (16 keys)

    for (int kt = 0; kt < 57; ++kt) {
        const int k0 = kt * 64;

        // ---- S = Q K^T (each wave: one 16x16 frag over K=512) ----
        v8f s = v8f_zero();
        int kc = min(k0 + 16 * nb + ln, LK_TOT - 1);
        const _Float16* krow = kp + (bL + kc) * (size_t)DM;
        const _Float16* qrow = &sQ[16 * mb + ln][0];

        // Prefetch next K-tile rows (lowers to global_prefetch_b8).
        {
            int kcn = min(k0 + 64 + 16 * nb + ln, LK_TOT - 1);
            const _Float16* krow_n = kp + (bL + kcn) * (size_t)DM;
            __builtin_prefetch(krow_n, 0, 1);
            __builtin_prefetch(krow_n + 256, 0, 1);
        }

#pragma unroll
        for (int ks = 0; ks < 16; ++ks) {
            v16h fa = load_frag(qrow + ks * 32, hi);
            v16h fb = load_frag(krow + ks * 32, hi);
            s = __builtin_amdgcn_wmma_f32_16x16x32_f16(
                false, fa, false, fb, (short)0, s, false, false);
        }
#pragma unroll
        for (int g = 0; g < 8; ++g)
            sS[16 * mb + hi * 8 + g][16 * nb + ln] = s[g];
        __syncthreads();

        // ---- online softmax: 8 threads per row, intra-wave shfl reduction ----
        {
            const int row = tid >> 3, sub = tid & 7;
            const int nvalid = min(64, LK_TOT - k0);
            float m_old = sM[row];
            float pm = -1e30f;
#pragma unroll
            for (int j = 0; j < 8; ++j) {
                int c = sub * 8 + j;
                float sv = (c < nvalid) ? sS[row][c] * scale : -1e30f;
                pm = fmaxf(pm, sv);
            }
            pm = fmaxf(pm, __shfl_xor(pm, 1));
            pm = fmaxf(pm, __shfl_xor(pm, 2));
            pm = fmaxf(pm, __shfl_xor(pm, 4));
            float m_new = fmaxf(m_old, pm);
            float alpha = __expf(m_old - m_new);
            float ps = 0.f;
#pragma unroll
            for (int j = 0; j < 8; ++j) {
                int c = sub * 8 + j;
                float p = (c < nvalid) ? __expf(sS[row][c] * scale - m_new) : 0.f;
                sP[row][c] = (_Float16)p;
                ps += p;
            }
            ps += __shfl_xor(ps, 1);
            ps += __shfl_xor(ps, 2);
            ps += __shfl_xor(ps, 4);
            if (sub == 0) {
                sL[row] = sL[row] * alpha + ps;
                sM[row] = m_new;
                sAlpha[row] = alpha;
            }
        }
        __syncthreads();

        // ---- rescale O, then O += P * V (wave owns 64 d-columns) ----
#pragma unroll
        for (int mi = 0; mi < 2; ++mi) {
#pragma unroll
            for (int g = 0; g < 8; ++g) {
                float al = sAlpha[16 * mi + hi * 8 + g];
#pragma unroll
                for (int ni = 0; ni < 4; ++ni) o[mi][ni][g] *= al;
            }
        }
#pragma unroll
        for (int ks = 0; ks < 2; ++ks) {
            v16h pa[2];
#pragma unroll
            for (int mi = 0; mi < 2; ++mi)
                pa[mi] = load_frag(&sP[16 * mi + ln][ks * 32], hi);
#pragma unroll
            for (int ni = 0; ni < 4; ++ni) {
                const int d = 64 * w + 16 * ni + ln;
                const _Float16* vrow = vpT + (size_t)d * M_TOT + bL;
                int l0 = k0 + ks * 32 + hi * 8;
                int l1 = l0 + 16;
                l0 = min(l0, LK_TOT - 8);  // clamped chunks are masked (P==0)
                l1 = min(l1, LK_TOT - 8);
                v16h vb = mk_frag(*(const v8h*)(vrow + l0), *(const v8h*)(vrow + l1));
#pragma unroll
                for (int mi = 0; mi < 2; ++mi)
                    o[mi][ni] = __builtin_amdgcn_wmma_f32_16x16x32_f16(
                        false, pa[mi], false, vb, (short)0, o[mi][ni], false, false);
            }
        }
    }

    // ---- normalize and store ----
#pragma unroll
    for (int mi = 0; mi < 2; ++mi) {
#pragma unroll
        for (int g = 0; g < 8; ++g) {
            const int row = 16 * mi + hi * 8 + g;
            const int gq  = q0 + row;
            if (gq < LQ_TOT) {
                float inv = 1.0f / sL[row];
#pragma unroll
                for (int ni = 0; ni < 4; ++ni) {
                    int d = 64 * w + 16 * ni + ln;
                    attn[(bL + gq) * (size_t)DM + d] = (_Float16)(o[mi][ni][g] * inv);
                }
            }
        }
    }
}

// ---------------------------------------------------------------------------
// In-place LayerNorm over rows of 512, one block (256 thr) per row.
// ---------------------------------------------------------------------------
__global__ void __launch_bounds__(256)
layernorm_inplace(float* __restrict__ out, const float* __restrict__ gamma,
                  const float* __restrict__ beta) {
    __shared__ float red[256];
    const int tid = threadIdx.x;
    float* p = out + (size_t)blockIdx.x * DM;
    float x0 = p[tid], x1 = p[tid + 256];

    red[tid] = x0 + x1;
    __syncthreads();
    for (int off = 128; off > 0; off >>= 1) {
        if (tid < off) red[tid] += red[tid + off];
        __syncthreads();
    }
    float mean = red[0] * (1.0f / DM);
    __syncthreads();

    float d0 = x0 - mean, d1 = x1 - mean;
    red[tid] = d0 * d0 + d1 * d1;
    __syncthreads();
    for (int off = 128; off > 0; off >>= 1) {
        if (tid < off) red[tid] += red[tid + off];
        __syncthreads();
    }
    float rstd = rsqrtf(red[0] * (1.0f / DM) + 1e-5f);

    p[tid]       = d0 * rstd * gamma[tid]       + beta[tid];
    p[tid + 256] = d1 * rstd * gamma[tid + 256] + beta[tid + 256];
}

// ---------------------------------------------------------------------------
extern "C" void kernel_launch(void* const* d_in, const int* in_sizes, int n_in,
                              void* d_out, int out_size, void* d_ws, size_t ws_size,
                              hipStream_t stream) {
    const float* q     = (const float*)d_in[0];
    const float* k     = (const float*)d_in[1];
    const float* v     = (const float*)d_in[2];
    const float* wqkv  = (const float*)d_in[3];
    const float* fcw   = (const float*)d_in[4];
    const float* fcb   = (const float*)d_in[5];
    const float* gamma = (const float*)d_in[6];
    const float* beta  = (const float*)d_in[7];
    float* out = (float*)d_out;

    char* ws = (char*)d_ws;
    const size_t SZ16 = (size_t)M_TOT * DM * sizeof(_Float16);  // 29,491,200 B
    _Float16* qh    = (_Float16*)(ws + 0 * SZ16);
    _Float16* kh    = (_Float16*)(ws + 1 * SZ16);
    _Float16* vh    = (_Float16*)(ws + 2 * SZ16);
    _Float16* qp    = (_Float16*)(ws + 3 * SZ16);
    _Float16* kp    = (_Float16*)(ws + 4 * SZ16);
    _Float16* vpT   = (_Float16*)(ws + 5 * SZ16);
    _Float16* w16   = (_Float16*)(ws + 6 * SZ16);
    _Float16* wfc16 = (_Float16*)(ws + 6 * SZ16 + (size_t)DM * DM * 2);
    _Float16* attn  = qh;  // qh no longer needed once projections are done

    // Stage 1: convert / transpose inputs to f16
    pack_f16<<<dim3((M_TOT * DM / 4) / 256), 256, 0, stream>>>(q, qh, M_TOT * DM / 4);
    pack_f16<<<dim3((DM * DM / 4) / 256), 256, 0, stream>>>(wqkv, w16, DM * DM / 4);
    pack_f16<<<dim3((DM * DM / 4) / 256), 256, 0, stream>>>(fcw, wfc16, DM * DM / 4);
    dim3 tg(113, 16, 8);
    transpose_pack<<<tg, 256, 0, stream>>>(k, kh);
    transpose_pack<<<tg, 256, 0, stream>>>(v, vh);

    // Stage 2: shared QKV projections (WMMA)
    dim3 gg(M_TOT / 128, 2);
    gemm512<0><<<gg, 256, 0, stream>>>(qh, w16, qp, nullptr, nullptr, nullptr);
    gemm512<0><<<gg, 256, 0, stream>>>(kh, w16, kp, nullptr, nullptr, nullptr);
    gemm512<1><<<gg, 256, 0, stream>>>(vh, w16, vpT, nullptr, nullptr, nullptr);

    // Stage 3: flash attention (WMMA + LDS + async staging + online softmax)
    dim3 fg(113, B_SZ);
    flash_attn<<<fg, 256, 0, stream>>>(qp, kp, vpT, attn);

    // Stage 4: fc projection + bias + residual -> d_out (f32)
    gemm512<2><<<gg, 256, 0, stream>>>(attn, wfc16, nullptr, out, fcb, q);

    // Stage 5: LayerNorm in place on d_out
    layernorm_inplace<<<dim3(M_TOT), 256, 0, stream>>>(out, gamma, beta);
}
